// DualTierMiras_42004780155430
// MI455X (gfx1250) — matrix operand order repaired and checked by reading, not compiled, not run on hardware
//
#include <hip/hip_runtime.h>
#include <hip/hip_bf16.h>
#include <math.h>

typedef __attribute__((ext_vector_type(16))) __bf16 v16bf;
typedef __attribute__((ext_vector_type(8)))  __bf16 v8bf;
typedef __attribute__((ext_vector_type(8)))  float  v8f;

#define BATCH 2048
#define DMODEL 256
#define HEADS 4
#define SLOTS 64
#define DH 64
#define PITCH 65   // LDS row pitch (floats): 65 % 64 == 1 -> column reads conflict-free

// ---------------------------------------------------------------------------
// f32 -> bf16 row-major convert (one-time operand pack). 4 elems/thread.
// ---------------------------------------------------------------------------
__global__ void cvt_bf16_kernel(const float* __restrict__ in,
                                __bf16* __restrict__ out) {
    size_t i = ((size_t)blockIdx.x * blockDim.x + threadIdx.x) * 4;
    float4 v = *(const float4*)(in + i);
    out[i + 0] = (__bf16)v.x;
    out[i + 1] = (__bf16)v.y;
    out[i + 2] = (__bf16)v.z;
    out[i + 3] = (__bf16)v.w;
}

// ---------------------------------------------------------------------------
// W[K=256,N=256] f32 -> Wt[N,K] bf16 (column-major pack for WMMA B-fragments).
// 32x32 LDS tile transpose; grid (8,8), 256 threads (32x8).
// ---------------------------------------------------------------------------
__global__ void transpose_bf16_kernel(const float* __restrict__ in,
                                      __bf16* __restrict__ out) {
    __shared__ float tile[32][33];
    const int k0 = blockIdx.x * 32;
    const int n0 = blockIdx.y * 32;
    const int tx = threadIdx.x & 31;
    const int ty = threadIdx.x >> 5;
    #pragma unroll
    for (int r = ty; r < 32; r += 8)
        tile[r][tx] = in[(size_t)(k0 + r) * DMODEL + n0 + tx];
    __syncthreads();
    #pragma unroll
    for (int r = ty; r < 32; r += 8)
        out[(size_t)(n0 + r) * DMODEL + k0 + tx] = (__bf16)tile[tx][r];
}

// ---------------------------------------------------------------------------
// WMMA GEMM: C[M,256] = A[M,256] * W[256,256] + bias, A bf16 row-major,
// Wt bf16 column-major. Block = 128 threads = 4 waves; wave computes a 16x64
// strip (4 accumulators). Per 32-wide K step: one v16bf A load + four v16bf
// B loads feed 4 back-to-back v_wmma_f32_16x16x32_bf16 (no converts, no
// per-element waits). A and B fragments use contiguous k = 16*half + s (wave32).
// ---------------------------------------------------------------------------
__global__ void wmma_gemm_bf16(const __bf16* __restrict__ A,
                               const __bf16* __restrict__ Wt,
                               const float* __restrict__ bias,
                               float* __restrict__ C,
                               int M, int N, int K) {
    const int lane = threadIdx.x & 31;
    const int wave = threadIdx.x >> 5;
    const int m0 = blockIdx.x * 16;
    const int n0 = wave * 64;            // 4 waves cover N = 256
    const int half = lane >> 4;          // K-half selector
    const int mr   = lane & 15;          // A row owned by this lane
    const int nc   = lane & 15;          // B/C column owned by this lane

    v8f acc[4] = {{}, {}, {}, {}};
    const __bf16* arow = A + (size_t)(m0 + mr) * K;

    for (int k0 = 0; k0 < K; k0 += 32) {
        // A fragment: lane's K values are 16 contiguous bf16 (k = 16*half + s)
        v16bf af = *(const v16bf*)(arow + k0 + half * 16);

        #pragma unroll
        for (int tile = 0; tile < 4; ++tile) {
            const int n = n0 + tile * 16 + nc;
            // B fragment: 16 contiguous bf16 from the packed column
            v16bf bf = *(const v16bf*)(Wt + (size_t)n * K + k0 + half * 16);
            acc[tile] = __builtin_amdgcn_wmma_f32_16x16x32_bf16(
                false, af, false, bf, (short)0, acc[tile], false, false);
        }
    }

    #pragma unroll
    for (int tile = 0; tile < 4; ++tile) {
        const int n = n0 + tile * 16 + nc;
        const float bv = bias[n];
        #pragma unroll
        for (int r = 0; r < 8; ++r) {
            int m = m0 + r + half * 8;   // VGPR r -> rows r / r+8 per lane half
            C[(size_t)m * N + n] = acc[tile][r] + bv;
        }
    }
}

// ---------------------------------------------------------------------------
// EMA surprise statistics + gate/mix. Block = 256 threads per batch row.
// ---------------------------------------------------------------------------
__global__ void stats_kernel(const float* __restrict__ hs,
                             const float* __restrict__ smean,
                             const float* __restrict__ svar,
                             const float* __restrict__ query,
                             const float* __restrict__ Wg,
                             const float* __restrict__ bg,
                             const float* __restrict__ mix_logit,
                             float* __restrict__ nmean,
                             float* __restrict__ nvar,
                             float* __restrict__ surprise,
                             float* __restrict__ mixv) {
    const int b = blockIdx.x;
    const int t = threadIdx.x;
    const size_t idx = (size_t)b * DMODEL + t;
    __shared__ float red[256];

    float h = hs[idx];
    float m = smean[idx];
    float v = svar[idx];
    nmean[idx] = 0.99f * m + 0.01f * h;
    float dmm = h - m;
    nvar[idx]  = 0.99f * v + 0.01f * dmm * dmm;

    red[t] = fabsf(dmm * rsqrtf(v + 1e-6f));
    __syncthreads();
    for (int ofs = 128; ofs > 0; ofs >>= 1) {
        if (t < ofs) red[t] += red[t + ofs];
        __syncthreads();
    }
    if (t == 0) {
        float z = red[0] * (1.f / (float)DMODEL);
        surprise[b] = 1.f / (1.f + expf(-(z - 2.0f)));   // 1/max(THR,0.1) = 2
    }
    __syncthreads();

    // gate = tanh(query[b] . Wg + bg) ; mix = sigmoid(mix_logit + gate)
    red[t] = query[idx] * Wg[t];
    __syncthreads();
    for (int ofs = 128; ofs > 0; ofs >>= 1) {
        if (t < ofs) red[t] += red[t + ofs];
        __syncthreads();
    }
    if (t == 0) {
        float gate = tanhf(red[0] + bg[0]);
        mixv[b] = 1.f / (1.f + expf(-(mix_logit[0] + gate)));
    }
}

// ---------------------------------------------------------------------------
// Fused dual-tier attention + surprise-gated circular-buffer write.
// One 256-thread block per (b,h). Each 16KB KV tile is read from HBM exactly
// once: staged to LDS (pitch-65 rows -> conflict-free column reads for AV)
// while the override-copy streams out with b128 stores. Emits `mixed` in bf16
// so the output GEMM consumes it directly.
// ---------------------------------------------------------------------------
__global__ void attn_scatter_kernel(const float* __restrict__ qh,
                                    const float* __restrict__ kh,
                                    const float* __restrict__ vh,
                                    const float* __restrict__ fk_in,
                                    const float* __restrict__ fv_in,
                                    const float* __restrict__ dk_in,
                                    const float* __restrict__ dv_in,
                                    const int* __restrict__ fptr,
                                    const int* __restrict__ dptr,
                                    const unsigned char* __restrict__ wmask,
                                    const float* __restrict__ surprise,
                                    const float* __restrict__ mixv,
                                    float* __restrict__ fk_out,
                                    float* __restrict__ fv_out,
                                    float* __restrict__ dk_out,
                                    float* __restrict__ dv_out,
                                    int* __restrict__ fp_out,
                                    int* __restrict__ dp_out,
                                    __bf16* __restrict__ mixed_bf) {
    const int bh = blockIdx.x;
    const int b = bh >> 2;
    const int h = bh & 3;
    const int t = threadIdx.x;

    __shared__ float kt[SLOTS * PITCH];
    __shared__ float vt[SLOTS * PITCH];
    __shared__ float qn_s[DH];
    __shared__ float attn_s[SLOTS];
    __shared__ float red[64];

    const bool wm = wmask[b] != 0;
    const bool dm = wm && (surprise[b] > 0.5f);
    const int fp = fptr[bh];
    const int dp = dptr[bh];
    const float mix = mixv[b];
    const size_t base = (size_t)bh * (SLOTS * DH);
    const float* krow = kh + (size_t)b * DMODEL + h * DH;   // replacement rows
    const float* vrow = vh + (size_t)b * DMODEL + h * DH;

    // normalize q head (threads 0..63)
    float qv = 0.f;
    if (t < 64) {
        qv = qh[(size_t)b * DMODEL + h * DH + t];
        red[t] = qv * qv;
    }
    __syncthreads();
    for (int ofs = 32; ofs > 0; ofs >>= 1) {
        if (t < ofs) red[t] += red[t + ofs];
        __syncthreads();
    }
    if (t < 64) qn_s[t] = qv / (sqrtf(red[0]) + 1e-8f);
    __syncthreads();

    float out_f = 0.f, out_d = 0.f;

    for (int tier = 0; tier < 2; ++tier) {
        const float* kin = tier ? dk_in : fk_in;
        const float* vin = tier ? dv_in : fv_in;
        float* kout = tier ? dk_out : fk_out;
        float* vout = tier ? dv_out : fv_out;
        const bool msk = tier ? dm : wm;
        const int  ptr = tier ? dp : fp;

        // single-pass: stage original tile to LDS + stream override-copy out
        #pragma unroll
        for (int it = 0; it < 4; ++it) {
            int vec = t + it * 256;          // 0..1023 float4s
            int s   = vec >> 4;              // 16 float4 per 64-float row
            int d4  = (vec & 15) * 4;
            float4 kd = ((const float4*)(kin + base))[vec];
            float4 vd = ((const float4*)(vin + base))[vec];
            float* kl = kt + s * PITCH + d4;
            float* vl = vt + s * PITCH + d4;
            kl[0] = kd.x; kl[1] = kd.y; kl[2] = kd.z; kl[3] = kd.w;
            vl[0] = vd.x; vl[1] = vd.y; vl[2] = vd.z; vl[3] = vd.w;
            if (msk && s == ptr) {
                kd.x = krow[d4]; kd.y = krow[d4 + 1]; kd.z = krow[d4 + 2]; kd.w = krow[d4 + 3];
                vd.x = vrow[d4]; vd.y = vrow[d4 + 1]; vd.z = vrow[d4 + 2]; vd.w = vrow[d4 + 3];
            }
            ((float4*)(kout + base))[vec] = kd;
            ((float4*)(vout + base))[vec] = vd;
        }
        __syncthreads();

        // cosine similarity: thread s = t (< 64), row read from LDS
        float sim = 0.f;
        if (t < 64) {
            float dot = 0.f, k2 = 0.f;
            #pragma unroll 8
            for (int d = 0; d < DH; ++d) {
                float kd = kt[t * PITCH + d];
                k2  += kd * kd;
                dot += kd * qn_s[d];
            }
            sim = dot / (sqrtf(k2) + 1e-8f);
            red[t] = sim;
        }
        __syncthreads();
        for (int ofs = 32; ofs > 0; ofs >>= 1) {
            if (t < ofs) red[t] = fmaxf(red[t], red[t + ofs]);
            __syncthreads();
        }
        float mx = red[0];
        __syncthreads();
        float e = 0.f;
        if (t < 64) { e = expf(sim - mx); red[t] = e; }
        __syncthreads();
        for (int ofs = 32; ofs > 0; ofs >>= 1) {
            if (t < ofs) red[t] += red[t + ofs];
            __syncthreads();
        }
        if (t < 64) attn_s[t] = e / red[0];
        __syncthreads();

        // AV: thread d = t (< 64); pitch-65 column reads are bank-conflict-free
        if (t < 64) {
            float o = 0.f;
            #pragma unroll 8
            for (int s = 0; s < SLOTS; ++s)
                o += attn_s[s] * vt[s * PITCH + t];
            if (tier) out_d = o; else out_f = o;
        }
        __syncthreads();   // LDS reused by next tier
    }

    if (t < 64)
        mixed_bf[(size_t)b * DMODEL + h * DH + t] =
            (__bf16)(mix * out_f + (1.f - mix) * out_d);
    if (t == 0) {
        fp_out[bh] = wm ? ((fp + 1) & (SLOTS - 1)) : fp;
        dp_out[bh] = dm ? ((dp + 1) & (SLOTS - 1)) : dp;
    }
}

// ---------------------------------------------------------------------------
extern "C" void kernel_launch(void* const* d_in, const int* in_sizes, int n_in,
                              void* d_out, int out_size, void* d_ws, size_t ws_size,
                              hipStream_t stream) {
    (void)in_sizes; (void)n_in; (void)out_size; (void)ws_size;

    const float* query  = (const float*)d_in[0];
    const float* wval   = (const float*)d_in[1];
    const unsigned char* wmask = (const unsigned char*)d_in[2];
    const float* fkeys  = (const float*)d_in[3];
    const float* fvals  = (const float*)d_in[4];
    const float* dkeys  = (const float*)d_in[5];
    const float* dvals  = (const float*)d_in[6];
    const int*   fptr   = (const int*)d_in[7];
    const int*   dptr   = (const int*)d_in[8];
    const float* smean  = (const float*)d_in[9];
    const float* svar   = (const float*)d_in[10];
    const float* Wq = (const float*)d_in[11]; const float* bq = (const float*)d_in[12];
    const float* Wk = (const float*)d_in[13]; const float* bk = (const float*)d_in[14];
    const float* Wv = (const float*)d_in[15]; const float* bv = (const float*)d_in[16];
    const float* Wo = (const float*)d_in[17]; const float* bo = (const float*)d_in[18];
    const float* Ws = (const float*)d_in[19]; const float* bs = (const float*)d_in[20];
    const float* Wg = (const float*)d_in[21]; const float* bg = (const float*)d_in[22];
    const float* mix_logit = (const float*)d_in[23];

    float* out = (float*)d_out;

    // d_out layout (flat, return order)
    const size_t BD   = (size_t)BATCH * DMODEL;                 // 524288
    const size_t BUF  = (size_t)BATCH * HEADS * SLOTS * DH;     // 33554432
    const size_t BH   = (size_t)BATCH * HEADS;                  // 8192
    const size_t WSZ  = (size_t)DMODEL * DMODEL;                // 65536
    const size_t o_fk = BD;
    const size_t o_fv = o_fk + BUF;
    const size_t o_dk = o_fv + BUF;
    const size_t o_dv = o_dk + BUF;
    const size_t o_fp = o_dv + BUF;
    const size_t o_dp = o_fp + BH;
    const size_t o_nm = o_dp + BH;
    const size_t o_nv = o_nm + BD;
    const size_t o_su = o_nv + BD;

    // workspace layout: f32 region, then bf16 region
    float* ws_f  = (float*)d_ws;
    float* qh    = ws_f;            // [B,D]  q projection (f32)
    float* kh    = ws_f + BD;       // [B,D]  k projection
    float* hs    = ws_f + 2 * BD;   // [B,D]  surprise projection
    float* vh    = ws_f + 3 * BD;   // [B,DV] write-value projection
    float* mixv  = ws_f + 4 * BD;   // [B]    mix gate

    __bf16* bf_base  = (__bf16*)(ws_f + 4 * BD + BATCH);
    __bf16* query_bf = bf_base;               // [B,D]  packed A
    __bf16* wval_bf  = bf_base + BD;          // [B,D]  packed A
    __bf16* mixed_bf = bf_base + 2 * BD;      // [B,DV] packed A (from attention)
    __bf16* Wq_t = bf_base + 3 * BD;          // [N,K] packed weights
    __bf16* Wk_t = Wq_t + WSZ;
    __bf16* Ws_t = Wk_t + WSZ;
    __bf16* Wv_t = Ws_t + WSZ;
    __bf16* Wo_t = Wv_t + WSZ;

    // --- one-time operand packs (bf16) ---
    cvt_bf16_kernel<<<BD / 1024, 256, 0, stream>>>(query, query_bf);
    cvt_bf16_kernel<<<BD / 1024, 256, 0, stream>>>(wval,  wval_bf);
    dim3 tgrid(8, 8, 1);
    transpose_bf16_kernel<<<tgrid, 256, 0, stream>>>(Wq, Wq_t);
    transpose_bf16_kernel<<<tgrid, 256, 0, stream>>>(Wk, Wk_t);
    transpose_bf16_kernel<<<tgrid, 256, 0, stream>>>(Ws, Ws_t);
    transpose_bf16_kernel<<<tgrid, 256, 0, stream>>>(Wv, Wv_t);
    transpose_bf16_kernel<<<tgrid, 256, 0, stream>>>(Wo, Wo_t);

    dim3 ggrid(BATCH / 16, 1, 1);   // 128 blocks
    dim3 gblk(128, 1, 1);           // 4 waves, each owning a 16x64 strip

    // Projections (WMMA GEMMs, 4 v_wmma per K-step per wave)
    wmma_gemm_bf16<<<ggrid, gblk, 0, stream>>>(query_bf, Wq_t, bq, qh, BATCH, DMODEL, DMODEL);
    wmma_gemm_bf16<<<ggrid, gblk, 0, stream>>>(query_bf, Wk_t, bk, kh, BATCH, DMODEL, DMODEL);
    wmma_gemm_bf16<<<ggrid, gblk, 0, stream>>>(query_bf, Ws_t, bs, hs, BATCH, DMODEL, DMODEL);
    wmma_gemm_bf16<<<ggrid, gblk, 0, stream>>>(wval_bf,  Wv_t, bv, vh, BATCH, DMODEL, DMODEL);

    // Surprise stats + gate/mix -> new_mean / new_var / surprise / mixv
    stats_kernel<<<BATCH, 256, 0, stream>>>(hs, smean, svar, query, Wg, bg, mix_logit,
                                            out + o_nm, out + o_nv, out + o_su, mixv);

    // Fused attention + gated circular-buffer write (single HBM pass per tile)
    attn_scatter_kernel<<<BATCH * HEADS, 256, 0, stream>>>(
        qh, kh, vh, fkeys, fvals, dkeys, dvals, fptr, dptr, wmask,
        out + o_su, mixv,
        out + o_fk, out + o_fv, out + o_dk, out + o_dv,
        (int*)(out + o_fp), (int*)(out + o_dp), mixed_bf);

    // Output projection (WMMA GEMM) -> d_out[0]
    wmma_gemm_bf16<<<ggrid, gblk, 0, stream>>>(mixed_bf, Wo_t, bo, out, BATCH, DMODEL, DMODEL);
}
